// Atom_embedding_MP_15942918603398
// MI455X (gfx1250) — matrix operand order repaired and checked by reading, not compiled, not run on hardware
//
#include <hip/hip_runtime.h>

// ---------------------------------------------------------------------------
// dMaSIF atom-embedding MP for gfx1250 (MI455X), wave32 + WMMA f16 16x16x32.
//   Kernel 1: brute-force masked KNN (k=16) with y tiled through LDS.
//   Kernel 2 (x3 launches, one per layer): per-point MLP message passing.
//     - TWO points per wave: each LDS B fragment feeds two wmmas
//       (16 wmma : 16 ds_load_b128 in the inner tile loop)
//     - explicit B double-buffering so ds_loads overlap the wmma pair
//     - lanes 0-15 own point0, lanes 16-31 own point1 in the epilogue
//     - f=256 (distance) column as VALU rank-1 update; h-col 256 as scalar
//     - W1/W2 pre-packed into LDS in B-fragment lane order
//     - 6 waves / workgroup, ~313KB LDS (<= 320KB WGP limit)
// ---------------------------------------------------------------------------

#define NXc 16000
#define NYc 6000
#define Dc  128
#define Kc  16
#define Hc  257   // 2*D + 1
#define Lc  3

typedef _Float16 v16h __attribute__((ext_vector_type(16)));
typedef _Float16 v8h  __attribute__((ext_vector_type(8)));
typedef float    v8f  __attribute__((ext_vector_type(8)));

// ---- LDS layout (bytes) for the MP kernel (dynamic LDS) -------------------
#define NT1 17                       // h output tiles (272 = pad of 257)
#define NT2 8                        // d output tiles (128)
#define KC1 8                        // dense k-dim chunks (256 feats)
#define TILE_HALVES (32*16)          // 512 halves = one 32x16 f16 B tile
#define W1F_BYTES (NT1*KC1*TILE_HALVES*2)  // 139264
#define W2F_BYTES (NT2*KC1*TILE_HALVES*2)  //  65536
#define B1S_OFF   (W1F_BYTES + W2F_BYTES)
#define B2S_OFF   (B1S_OFF + 272*4)
#define GS_OFF    (B2S_OFF + 128*4)
#define BS_OFF    (GS_OFF  + 128*4)
#define W1C_OFF   (BS_OFF  + 128*4)        // W1[:,256]  (272 floats)
#define W2C_OFF   (W1C_OFF + 272*4)        // W2[:,256]  (128 floats)
#define FIXED_BYTES (W2C_OFF + 128*4)      // 209024
#define HA_STRIDE  264                     // halves per row of h staging
#define HA_BYTES  (16*HA_STRIDE*2)         // 8448
#define NBR_STRIDE 136                     // halves per row of nbr staging
#define PW_BYTES  (2*HA_BYTES + 2*128*2)   // 17408 per wave (2x hA + 2x pe)
#define WAVES 6
#define MP_THREADS (WAVES*32)              // 192
#define SMEM_BYTES (FIXED_BYTES + WAVES*PW_BYTES)  // 313472 <= 320KB
#define MP_BLOCKS 128
#define NPAIRS (NXc/2)

__device__ __forceinline__ float lrelu(float v) {
    return v >= 0.f ? v : 0.2f * v;
}

// Build a v16h fragment from two 8-half (16B) runs in LDS.
__device__ __forceinline__ v16h combine(const _Float16* p0, const _Float16* p1) {
    v8h lo = *(const v8h*)p0;
    v8h hi = *(const v8h*)p1;
    v16h r;
#pragma unroll
    for (int i = 0; i < 8; ++i) { r[i] = lo[i]; r[i + 8] = hi[i]; }
    return r;
}

// Load the kc-th 32x16 B tile fragment for this lane (two ds_load_b128).
__device__ __forceinline__ v16h loadB(const _Float16* bp, int kc) {
    return combine(bp + kc * TILE_HALVES, bp + kc * TILE_HALVES + 8);
}

// ---------------------------------------------------------------------------
// KNN: one thread per query point, y staged through LDS in tiles.
// ---------------------------------------------------------------------------
#define TY 2048
__global__ void __launch_bounds__(256)
knn_kernel(const float* __restrict__ x, const float* __restrict__ y,
           const int* __restrict__ xb, const int* __restrict__ yb,
           int* __restrict__ idx_out, float* __restrict__ dist_out)
{
    __shared__ float sy[TY * 3];
    __shared__ int   sb[TY];

    const int tid = threadIdx.x;
    const int n   = blockIdx.x * 256 + tid;
    const bool valid = n < NXc;
    const int nn = valid ? n : 0;

    const float px = x[nn * 3 + 0];
    const float py = x[nn * 3 + 1];
    const float pz = x[nn * 3 + 2];
    const int   pb = xb[nn];

    float bd[Kc];
    int   bi[Kc];
#pragma unroll
    for (int s = 0; s < Kc; ++s) { bd[s] = 3.0e38f; bi[s] = 0; }

    for (int base = 0; base < NYc; base += TY) {
        const int cnt = (NYc - base) < TY ? (NYc - base) : TY;
        __syncthreads();
        for (int e = tid; e < cnt; e += 256) {
            sy[e * 3 + 0] = y[(base + e) * 3 + 0];
            sy[e * 3 + 1] = y[(base + e) * 3 + 1];
            sy[e * 3 + 2] = y[(base + e) * 3 + 2];
            sb[e]         = yb[base + e];
        }
        __syncthreads();
        for (int t = 0; t < cnt; ++t) {
            const float dx = px - sy[t * 3 + 0];
            const float dy = py - sy[t * 3 + 1];
            const float dz = pz - sy[t * 3 + 2];
            float d2 = dx * dx + dy * dy + dz * dz;
            d2 = (sb[t] == pb) ? d2 : 1.0e6f;   // cross-batch penalty
            if (d2 < bd[Kc - 1]) {
                const int gi = base + t;
#pragma unroll
                for (int s = Kc - 1; s >= 0; --s) {
                    const bool lt = d2 < bd[s];
                    const bool at = (s == 0) || (d2 >= bd[s - 1]);
                    const float nb = lt ? (at ? d2 : bd[s - 1]) : bd[s];
                    const int   ni = lt ? (at ? gi : bi[s - 1]) : bi[s];
                    bd[s] = nb; bi[s] = ni;
                }
            }
        }
    }
    if (valid) {
#pragma unroll
        for (int s = 0; s < Kc; ++s) {
            idx_out[n * Kc + s]  = bi[s];
            dist_out[n * Kc + s] = bd[s];
        }
    }
}

// ---------------------------------------------------------------------------
// One MP layer. Two points per wave; weights pre-packed in LDS B-fragment
// order; A fragments built per documented 16-bit A layout.
// ---------------------------------------------------------------------------
__global__ void __launch_bounds__(MP_THREADS)
mp_layer_kernel(const float* __restrict__ W1, const float* __restrict__ b1,
                const float* __restrict__ W2, const float* __restrict__ b2,
                const float* __restrict__ gamma_, const float* __restrict__ beta_,
                const float* __restrict__ yat, const int* __restrict__ idx,
                const float* __restrict__ dist, const float* __restrict__ pe_in,
                float* __restrict__ pe_out, int layer, int first)
{
    extern __shared__ char smem[];
    _Float16* w1f = (_Float16*)smem;
    _Float16* w2f = (_Float16*)(smem + W1F_BYTES);
    float* b1s = (float*)(smem + B1S_OFF);
    float* b2s = (float*)(smem + B2S_OFF);
    float* gs  = (float*)(smem + GS_OFF);
    float* bs  = (float*)(smem + BS_OFF);
    float* w1c = (float*)(smem + W1C_OFF);   // W1[n][256]
    float* w2c = (float*)(smem + W2C_OFF);   // W2[d][256]

    const int tid  = threadIdx.x;
    const int lane = tid & 31;
    const int wv   = tid >> 5;
    const int c    = lane & 15;     // fragment column (N), and A row index (M)
    const int hiL  = lane >> 4;     // lane-half selector / point selector

    char* wb = smem + FIXED_BYTES + wv * PW_BYTES;
    _Float16* hA0  = (_Float16*)wb;
    _Float16* hA1  = (_Float16*)(wb + HA_BYTES);
    _Float16* peL0 = (_Float16*)(wb + 2 * HA_BYTES);
    _Float16* peL1 = (_Float16*)(wb + 2 * HA_BYTES + 256);

    // ---- stage layer weights into LDS, pre-packed in B-fragment order ----
    // B layout (32x16 f16): lane n=L%16; halves j: k = (j&7) + (L>=16 ? 8 : 0) + (j>=8 ? 16 : 0)
    const float* W1l = W1 + (size_t)layer * Hc * Hc;
    for (int e = tid; e < NT1 * KC1 * TILE_HALVES; e += MP_THREADS) {
        const int j  = e & 15;
        const int ln = (e >> 4) & 31;
        const int t  = e >> 9;
        const int kc = t & 7;
        const int nt = t >> 3;
        const int h  = nt * 16 + (ln & 15);
        const int f  = kc * 32 + (j & 7) + ((ln & 16) ? 8 : 0) + ((j & 8) ? 16 : 0);   // f < 256 always
        float v = (h < Hc) ? W1l[h * Hc + f] : 0.f;
        w1f[e] = (_Float16)v;
    }
    const float* W2l = W2 + (size_t)layer * Dc * Hc;
    for (int e = tid; e < NT2 * KC1 * TILE_HALVES; e += MP_THREADS) {
        const int j  = e & 15;
        const int ln = (e >> 4) & 31;
        const int t  = e >> 9;
        const int kc = t & 7;
        const int nt = t >> 3;
        const int d  = nt * 16 + (ln & 15);
        const int f  = kc * 32 + (j & 7) + ((ln & 16) ? 8 : 0) + ((j & 8) ? 16 : 0);   // f < 256 always
        w2f[e] = (_Float16)W2l[d * Hc + f];
    }
    for (int e = tid; e < 272; e += MP_THREADS) {
        b1s[e] = (e < Hc) ? b1[layer * Hc + e] : 0.f;
        w1c[e] = (e < Hc) ? W1l[e * Hc + 256] : 0.f;
    }
    for (int e = tid; e < 128; e += MP_THREADS) {
        b2s[e] = b2[layer * Dc + e];
        gs[e]  = gamma_[layer * Dc + e];
        bs[e]  = beta_[layer * Dc + e];
        w2c[e] = W2l[e * Hc + 256];
    }
    __syncthreads();

    const int strideP = gridDim.x * WAVES;          // pairs per sweep
    const int gp      = blockIdx.x * WAVES + wv;
    const int iters   = (NPAIRS + strideP - 1) / strideP;

    for (int it = 0; it < iters; ++it) {
        const int pair = gp + it * strideP;
        const bool pvalid = pair < NPAIRS;
        const int pp = pvalid ? pair : 0;
        const int n0 = 2 * pp;
        const int n1 = 2 * pp + 1;
        const int nMine = hiL ? n1 : n0;   // this half-lane's point

        // ---- per-point neighbor data: lower half holds p0, upper holds p1
        const int   myidx  = idx[nMine * Kc + c];
        const float mydist = fminf(dist[nMine * Kc + c], 60000.f);

        // dist[m] per output row (m = r + 8*hiL) for both points
        float dv0[8], dv1[8];
#pragma unroll
        for (int r = 0; r < 8; ++r) {
            dv0[r] = __shfl(mydist, r + 8 * hiL, 32);
            dv1[r] = __shfl(mydist, 16 + r + 8 * hiL, 32);
        }

        // stage nbr = y_atomtypes[idx] as f16, rows padded to NBR_STRIDE
        for (int e = lane; e < Kc * Dc; e += 32) {
            const int m = e >> 7;
            const int f = e & 127;
            const int g0 = __shfl(myidx, m, 32);
            const int g1 = __shfl(myidx, m + 16, 32);
            hA0[m * NBR_STRIDE + f] = (_Float16)yat[g0 * Dc + f];
            hA1[m * NBR_STRIDE + f] = (_Float16)yat[g1 * Dc + f];
        }
        asm volatile("s_wait_dscnt 0" ::: "memory");

        // A layout (16x32 f16): lane m=L%16; two 8-half runs per chunk.
        v16h aN0[4], aN1[4];
#pragma unroll
        for (int cc = 0; cc < 4; ++cc) {
            const int off = c * NBR_STRIDE + cc * 32 + (hiL ? 8 : 0);
            aN0[cc] = combine(hA0 + off, hA0 + off + 16);
            aN1[cc] = combine(hA1 + off, hA1 + off + 16);
        }

        // ---- point embedding (persistent residual, f32 in registers) ----
        float pev[8];
#pragma unroll
        for (int t = 0; t < 8; ++t)
            pev[t] = first ? 1.0f : pe_in[(size_t)nMine * Dc + t * 16 + c];
        {
            _Float16* peMine = hiL ? peL1 : peL0;
#pragma unroll
            for (int t = 0; t < 8; ++t) peMine[t * 16 + c] = (_Float16)pev[t];
        }
        asm volatile("s_wait_dscnt 0" ::: "memory");

        v16h aP0[4], aP1[4];
#pragma unroll
        for (int cc = 0; cc < 4; ++cc) {
            const int off = cc * 32 + (hiL ? 8 : 0);
            aP0[cc] = combine(peL0 + off, peL0 + off + 16);
            aP1[cc] = combine(peL1 + off, peL1 + off + 16);
        }

        // ---- GEMM1: h = lrelu(feats * W1^T + dist x W1[:,256] + b1) ------
        for (int nt = 0; nt < 16; ++nt) {
            v8f acc0 = {}, acc1 = {};
            const _Float16* bp = w1f + (nt * KC1) * TILE_HALVES + lane * 16;
            v16h bcur = loadB(bp, 0);
#pragma unroll
            for (int kc = 0; kc < KC1; ++kc) {
                v16h bnext = (kc < KC1 - 1) ? loadB(bp, kc + 1) : bcur;
                v16h a0 = (kc < 4) ? aP0[kc] : aN0[kc - 4];
                v16h a1 = (kc < 4) ? aP1[kc] : aN1[kc - 4];
                acc0 = __builtin_amdgcn_wmma_f32_16x16x32_f16(
                    false, a0, false, bcur, (short)0, acc0, false, false);
                acc1 = __builtin_amdgcn_wmma_f32_16x16x32_f16(
                    false, a1, false, bcur, (short)0, acc1, false, false);
                bcur = bnext;
            }
            const int ng = nt * 16 + c;
            const float bb  = b1s[ng];
            const float w1v = w1c[ng];
#pragma unroll
            for (int r = 0; r < 8; ++r) {
                const int row = (r + 8 * hiL) * HA_STRIDE + ng;
                hA0[row] = (_Float16)lrelu(acc0[r] + bb + dv0[r] * w1v);
                hA1[row] = (_Float16)lrelu(acc1[r] + bb + dv1[r] * w1v);
            }
        }
        // h column 256 (tile nt=16, only n==256 live): scalar k-sums
        float sumhMine;
        {
            v8f acc0 = {}, acc1 = {};
            const _Float16* bp = w1f + (16 * KC1) * TILE_HALVES + lane * 16;
            v16h bcur = loadB(bp, 0);
#pragma unroll
            for (int kc = 0; kc < KC1; ++kc) {
                v16h bnext = (kc < KC1 - 1) ? loadB(bp, kc + 1) : bcur;
                v16h a0 = (kc < 4) ? aP0[kc] : aN0[kc - 4];
                v16h a1 = (kc < 4) ? aP1[kc] : aN1[kc - 4];
                acc0 = __builtin_amdgcn_wmma_f32_16x16x32_f16(
                    false, a0, false, bcur, (short)0, acc0, false, false);
                acc1 = __builtin_amdgcn_wmma_f32_16x16x32_f16(
                    false, a1, false, bcur, (short)0, acc1, false, false);
                bcur = bnext;
            }
            const int ng = 256 + c;
            const float bb  = b1s[ng];
            const float w1v = w1c[ng];
            float s0 = 0.f, s1 = 0.f;
#pragma unroll
            for (int r = 0; r < 8; ++r) {
                s0 += lrelu(acc0[r] + bb + dv0[r] * w1v);
                s1 += lrelu(acc1[r] + bb + dv1[r] * w1v);
            }
            s0 = (c == 0) ? s0 : 0.f;        // only column 256 is real
            s1 = (c == 0) ? s1 : 0.f;
#pragma unroll
            for (int off = 1; off < 32; off <<= 1) {
                s0 += __shfl_xor(s0, off, 32);
                s1 += __shfl_xor(s1, off, 32);
            }
            sumhMine = hiL ? s1 : s0;        // Σ_m h[m][256] of my point
        }
        asm volatile("s_wait_dscnt 0" ::: "memory");

        // reload h as A fragments (cols 0..255)
        v16h hF0[KC1], hF1[KC1];
#pragma unroll
        for (int kc = 0; kc < KC1; ++kc) {
            const int off = c * HA_STRIDE + kc * 32 + (hiL ? 8 : 0);
            hF0[kc] = combine(hA0 + off, hA0 + off + 16);
            hF1[kc] = combine(hA1 + off, hA1 + off + 16);
        }

        // ---- GEMM2: msg = h * W2^T, summed over neighbors ----------------
        float msgv[8];                        // colsums of MY point
#pragma unroll
        for (int dt = 0; dt < NT2; ++dt) {
            v8f acc0 = {}, acc1 = {};
            const _Float16* bp = w2f + (dt * KC1) * TILE_HALVES + lane * 16;
            v16h bcur = loadB(bp, 0);
#pragma unroll
            for (int kc = 0; kc < KC1; ++kc) {
                v16h bnext = (kc < KC1 - 1) ? loadB(bp, kc + 1) : bcur;
                acc0 = __builtin_amdgcn_wmma_f32_16x16x32_f16(
                    false, hF0[kc], false, bcur, (short)0, acc0, false, false);
                acc1 = __builtin_amdgcn_wmma_f32_16x16x32_f16(
                    false, hF1[kc], false, bcur, (short)0, acc1, false, false);
                bcur = bnext;
            }
            float s0 = 0.f, s1 = 0.f;
#pragma unroll
            for (int r = 0; r < 8; ++r) { s0 += acc0[r]; s1 += acc1[r]; }
            s0 += __shfl_xor(s0, 16, 32);     // fold halves -> all m
            s1 += __shfl_xor(s1, 16, 32);
            const int dg = dt * 16 + c;
            const float sMine = hiL ? s1 : s0;
            msgv[dt] = sMine + 16.0f * b2s[dg] + sumhMine * w2c[dg];
        }

        // ---- GroupNorm(1, D) per half-wave (each half owns its point) ----
        float sm = 0.f, sq = 0.f;
#pragma unroll
        for (int t = 0; t < 8; ++t) { sm += msgv[t]; sq += msgv[t] * msgv[t]; }
#pragma unroll
        for (int off = 1; off < 16; off <<= 1) {   // reduce within half-wave
            sm += __shfl_xor(sm, off, 32);
            sq += __shfl_xor(sq, off, 32);
        }
        const float mu  = sm * (1.0f / 128.0f);
        const float var = sq * (1.0f / 128.0f) - mu * mu;
        const float inv = rsqrtf(var + 1e-5f);
#pragma unroll
        for (int t = 0; t < 8; ++t) {
            const int d = t * 16 + c;
            const float nv = (msgv[t] - mu) * inv * gs[d] + bs[d];
            pev[t] += lrelu(nv);
        }
        if (pvalid) {
#pragma unroll
            for (int t = 0; t < 8; ++t)
                pe_out[(size_t)nMine * Dc + t * 16 + c] = pev[t];
        }
    }
}

// ---------------------------------------------------------------------------
extern "C" void kernel_launch(void* const* d_in, const int* in_sizes, int n_in,
                              void* d_out, int out_size, void* d_ws, size_t ws_size,
                              hipStream_t stream) {
    const float* x     = (const float*)d_in[0];
    const float* y     = (const float*)d_in[1];
    const float* yat   = (const float*)d_in[2];
    const int*   xb    = (const int*)d_in[3];
    const int*   yb    = (const int*)d_in[4];
    const float* W1    = (const float*)d_in[5];
    const float* b1    = (const float*)d_in[6];
    const float* W2    = (const float*)d_in[7];
    const float* b2    = (const float*)d_in[8];
    const float* gamma = (const float*)d_in[9];
    const float* beta  = (const float*)d_in[10];
    float* out = (float*)d_out;

    char* ws = (char*)d_ws;
    int*   idx_ws  = (int*)ws;                                   // NX*K ints
    float* dist_ws = (float*)(ws + (size_t)NXc * Kc * 4);        // NX*K floats
    float* peA     = (float*)(ws + (size_t)NXc * Kc * 8);        // NX*D floats
    float* peB     = peA + (size_t)NXc * Dc;                     // NX*D floats

    knn_kernel<<<(NXc + 255) / 256, 256, 0, stream>>>(x, y, xb, yb, idx_ws, dist_ws);

    mp_layer_kernel<<<MP_BLOCKS, MP_THREADS, SMEM_BYTES, stream>>>(
        W1, b1, W2, b2, gamma, beta, yat, idx_ws, dist_ws, peA, peA, 0, 1);
    mp_layer_kernel<<<MP_BLOCKS, MP_THREADS, SMEM_BYTES, stream>>>(
        W1, b1, W2, b2, gamma, beta, yat, idx_ws, dist_ws, peA, peB, 1, 0);
    mp_layer_kernel<<<MP_BLOCKS, MP_THREADS, SMEM_BYTES, stream>>>(
        W1, b1, W2, b2, gamma, beta, yat, idx_ws, dist_ws, peB, out, 2, 0);
}
